// MambaBlock_68161130988021
// MI455X (gfx1250) — compile-verified, matrix operand order checked by reading, not run on hardware
//
#include <hip/hip_runtime.h>
#include <hip/hip_bf16.h>

typedef __bf16 bf16_t;
typedef __attribute__((ext_vector_type(8)))  bf16_t v8bf;
typedef __attribute__((ext_vector_type(16))) bf16_t v16bf;
typedef __attribute__((ext_vector_type(8)))  float  v8f;

#define DM   768
#define DI   1536
#define LSEQ 2048
#define NB   2
#define MTOT (NB * LSEQ)   // 4096
#define RMS_EPS 1.1920928955078125e-07f

// LDS slab geometry for the cooperative GEMM
#define BM 128
#define BN 128
#define BK 32
#define LPAD 40            // padded row stride (elements) => 80B rows, 16B aligned chunks

// ---------------------------------------------------------------- cast f32->bf16
__global__ void mamba_cast_bf16(const float* __restrict__ s, bf16_t* __restrict__ d, int n) {
    int i = blockIdx.x * blockDim.x + threadIdx.x;
    if (i < n) d[i] = (bf16_t)s[i];
}

// ---------------------------------------------------------------- RMSNorm -> bf16
__global__ void mamba_rmsnorm(const float* __restrict__ x, const float* __restrict__ w,
                              bf16_t* __restrict__ xn) {
    int row = blockIdx.x;                       // 0..MTOT-1
    const float* xr = x + (size_t)row * DM;
    __shared__ float red[8];
    float ss = 0.f;
    for (int j = threadIdx.x; j < DM; j += blockDim.x) { float v = xr[j]; ss += v * v; }
    #pragma unroll
    for (int off = 16; off > 0; off >>= 1) ss += __shfl_down(ss, off, 32);
    if ((threadIdx.x & 31) == 0) red[threadIdx.x >> 5] = ss;
    __syncthreads();
    if (threadIdx.x == 0) {
        float t = 0.f;
        for (int i = 0; i < 8; ++i) t += red[i];
        red[0] = rsqrtf(t / (float)DM + RMS_EPS);
    }
    __syncthreads();
    float scale = red[0];
    for (int j = threadIdx.x; j < DM; j += blockDim.x)
        xn[(size_t)row * DM + j] = (bf16_t)(xr[j] * scale * w[j]);
}

// ---------------------------------------------------------------- async copy helpers
__device__ __forceinline__ void async_ld16(bf16_t* lds, const bf16_t* g) {
    uint32_t loff = (uint32_t)(size_t)lds;      // low 32 bits of generic shared ptr = LDS offset
    asm volatile("global_load_async_to_lds_b128 %0, %1, off"
                 :: "v"(loff), "v"(g) : "memory");
}

// stage ROWS x 32 bf16 slab (rows clamped to rlimit) into LDS with LPAD row stride
template <int ROWS>
__device__ __forceinline__ void stage_slab(bf16_t* __restrict__ lds,
                                           const bf16_t* __restrict__ gbase,
                                           int ld, int k0, int rlimit) {
    constexpr int CHUNKS = ROWS * 4;            // 16B chunks
    int tid = threadIdx.x;
    #pragma unroll
    for (int j = 0; j < CHUNKS / 256; ++j) {
        int c   = tid + j * 256;
        int row = c >> 2, sub = c & 3;
        int gr  = row < rlimit ? row : rlimit - 1;
        async_ld16(lds + row * LPAD + sub * 8,
                   gbase + (size_t)gr * ld + k0 + sub * 8);
    }
}

// ---------------------------------------------------------------- WMMA fragment from LDS
__device__ __forceinline__ v16bf mamba_frag(const bf16_t* __restrict__ base, int ld) {
    int lane = threadIdx.x & 31;
    const bf16_t* p = base + (size_t)(lane & 15) * ld + ((lane >> 4) << 3);
    v8bf lo = *(const v8bf*)(p);
    v8bf hi = *(const v8bf*)(p + 16);
    v16bf f;
    #pragma unroll
    for (int i = 0; i < 8; ++i) { f[i] = lo[i]; f[i + 8] = hi[i]; }
    return f;
}

// ---------------------------------------------------------------- cooperative WMMA bf16 GEMM
// C[M,N](f32,ldc) = A[M,K](bf16,lda) * Bw[N,K](bf16,ldb)^T (+ addsrc[M,N])
// block = 256 thr = 8 waves (4x2), macro-tile 128x128, wave tile 32x64, BK=32,
// async double-buffered LDS (40KB).
__global__ void __launch_bounds__(256)
mamba_gemm_wmma(const bf16_t* __restrict__ A, int lda,
                const bf16_t* __restrict__ Bw, int ldb,
                float* __restrict__ C, int ldc,
                int M, int N, int K,
                const float* __restrict__ addsrc) {
    __shared__ __align__(16) bf16_t sA[2][BM * LPAD];
    __shared__ __align__(16) bf16_t sB[2][BN * LPAD];

    int nbn = (N + BN - 1) / BN;
    int bm  = blockIdx.x / nbn;
    int bn  = blockIdx.x % nbn;

    const bf16_t* Ablk = A  + (size_t)(bm * BM) * lda;
    const bf16_t* Bblk = Bw + (size_t)(bn * BN) * ldb;
    int brows = N - bn * BN;                    // valid B rows in this block (may be < BN)

    int w  = threadIdx.x >> 5;
    int wm = w >> 1;                            // 0..3  (M direction, 32 rows each)
    int wn = w & 1;                             // 0..1  (N direction, 64 cols each)
    const bf16_t* la = &sA[0][(wm * 32) * LPAD];
    const bf16_t* lb = &sB[0][(wn * 64) * LPAD];
    int bufstepA = BM * LPAD;                   // element stride between double buffers
    int bufstepB = BN * LPAD;

    // prologue: stage k=0 into buffer 0 (4 async b128 per thread)
    stage_slab<BM>(sA[0], Ablk, lda, 0, BM);
    stage_slab<BN>(sB[0], Bblk, ldb, 0, brows);

    v8f acc[2][4] = {};
    int cur = 0;
    for (int k0 = 0; k0 < K; k0 += BK) {
        bool has_next = (k0 + BK) < K;
        if (has_next) {
            stage_slab<BM>(sA[cur ^ 1], Ablk, lda, k0 + BK, BM);
            stage_slab<BN>(sB[cur ^ 1], Bblk, ldb, k0 + BK, brows);
            if (k0 + 2 * BK < K) {              // near-scope stream-ahead
                __builtin_prefetch(Ablk + (size_t)(threadIdx.x >> 1) * lda + k0 + 2 * BK, 0, 3);
                __builtin_prefetch(Bblk + (size_t)(threadIdx.x >> 1) * ldb + k0 + 2 * BK, 0, 3);
            }
            asm volatile("s_wait_asynccnt 0x4" ::: "memory");  // current buffer complete
        } else {
            asm volatile("s_wait_asynccnt 0x0" ::: "memory");
        }
        __syncthreads();

        int oa = cur * bufstepA, ob = cur * bufstepB;
        v16bf a0 = mamba_frag(la + oa, LPAD);
        v16bf a1 = mamba_frag(la + oa + 16 * LPAD, LPAD);
        v16bf b0 = mamba_frag(lb + ob, LPAD);
        v16bf b1 = mamba_frag(lb + ob + 16 * LPAD, LPAD);
        v16bf b2 = mamba_frag(lb + ob + 32 * LPAD, LPAD);
        v16bf b3 = mamba_frag(lb + ob + 48 * LPAD, LPAD);
        acc[0][0] = __builtin_amdgcn_wmma_f32_16x16x32_bf16(false, a0, false, b0, (short)0, acc[0][0], false, false);
        acc[0][1] = __builtin_amdgcn_wmma_f32_16x16x32_bf16(false, a0, false, b1, (short)0, acc[0][1], false, false);
        acc[0][2] = __builtin_amdgcn_wmma_f32_16x16x32_bf16(false, a0, false, b2, (short)0, acc[0][2], false, false);
        acc[0][3] = __builtin_amdgcn_wmma_f32_16x16x32_bf16(false, a0, false, b3, (short)0, acc[0][3], false, false);
        acc[1][0] = __builtin_amdgcn_wmma_f32_16x16x32_bf16(false, a1, false, b0, (short)0, acc[1][0], false, false);
        acc[1][1] = __builtin_amdgcn_wmma_f32_16x16x32_bf16(false, a1, false, b1, (short)0, acc[1][1], false, false);
        acc[1][2] = __builtin_amdgcn_wmma_f32_16x16x32_bf16(false, a1, false, b2, (short)0, acc[1][2], false, false);
        acc[1][3] = __builtin_amdgcn_wmma_f32_16x16x32_bf16(false, a1, false, b3, (short)0, acc[1][3], false, false);

        __syncthreads();                        // all waves done with cur before restage
        cur ^= 1;
    }

    int lane = threadIdx.x & 31;
    int col  = lane & 15;
    int rb   = (lane >> 4) << 3;
    int row0 = bm * BM + wm * 32;
    int col0 = bn * BN + wn * 64;
    #pragma unroll
    for (int i = 0; i < 2; ++i) {
        #pragma unroll
        for (int j = 0; j < 4; ++j) {
            int n = col0 + j * 16 + col;
            if (n < N) {
                #pragma unroll
                for (int r = 0; r < 8; ++r) {
                    int m = row0 + i * 16 + rb + r;
                    float v = acc[i][j][r];
                    if (addsrc) v += addsrc[(size_t)m * ldc + n];
                    C[(size_t)m * ldc + n] = v;
                }
            }
        }
    }
}

// ---------------------------------------------------------------- causal depthwise conv(K=4)+SiLU
__global__ void mamba_conv_silu(const float* __restrict__ xz,
                                const float* __restrict__ cw, const float* __restrict__ cb,
                                float* __restrict__ u, bf16_t* __restrict__ ubf) {
    int idx = blockIdx.x * blockDim.x + threadIdx.x;
    if (idx >= MTOT * DI) return;
    int c = idx % DI;
    int l = (idx / DI) % LSEQ;
    int b = idx / (DI * LSEQ);
    const float* base = xz + (size_t)b * LSEQ * (2 * DI) + c;
    float acc = cb[c];
    #pragma unroll
    for (int k = 0; k < 4; ++k) {
        int lt = l - 3 + k;
        if (lt >= 0) acc += cw[c * 4 + k] * base[(size_t)lt * (2 * DI)];
    }
    float s = acc / (1.f + __expf(-acc));
    u[idx]   = s;
    ubf[idx] = (bf16_t)s;
}

// ---------------------------------------------------------------- selective scan + gating
__global__ void mamba_scan(const float* __restrict__ dtraw, const float* __restrict__ dtb,
                           const float* __restrict__ u, const float* __restrict__ xz,
                           const float* __restrict__ bc, const float* __restrict__ A_log,
                           const float* __restrict__ Dskip, bf16_t* __restrict__ ygated) {
    const int bpb = DI / 256;                    // blocks per batch
    int b = blockIdx.x / bpb;
    int c = (blockIdx.x % bpb) * 256 + threadIdx.x;

    float Aneg[16], h[16];
    #pragma unroll
    for (int n = 0; n < 16; ++n) { Aneg[n] = -__expf(A_log[c * 16 + n]); h[n] = 0.f; }
    float bias = dtb[c];
    float dsk  = Dskip[c];

    __shared__ float bcsh[32];
    for (int t = 0; t < LSEQ; ++t) {
        size_t row = (size_t)b * LSEQ + t;
        if (threadIdx.x < 32) bcsh[threadIdx.x] = bc[row * 32 + threadIdx.x];
        __syncthreads();
        float dr = dtraw[row * DI + c] + bias;
        float dl = (dr > 20.f) ? dr : log1pf(__expf(dr));     // softplus
        float ut = u[row * DI + c];
        float zv = xz[row * (2 * DI) + DI + c];
        float du = dl * ut;
        float y  = 0.f;
        #pragma unroll
        for (int n = 0; n < 16; ++n) {
            float dA = __expf(dl * Aneg[n]);
            h[n] = dA * h[n] + du * bcsh[n];
            y   += h[n] * bcsh[16 + n];
        }
        y += ut * dsk;
        float sz = zv / (1.f + __expf(-zv));                  // silu(z)
        ygated[row * DI + c] = (bf16_t)(y * sz);
        __syncthreads();
    }
}

// ---------------------------------------------------------------- launcher
extern "C" void kernel_launch(void* const* d_in, const int* in_sizes, int n_in,
                              void* d_out, int out_size, void* d_ws, size_t ws_size,
                              hipStream_t stream) {
    const float* x         = (const float*)d_in[0];
    const float* norm_w    = (const float*)d_in[1];
    const float* in_proj_w = (const float*)d_in[2];
    const float* conv_w    = (const float*)d_in[3];
    const float* conv_b    = (const float*)d_in[4];
    const float* x_proj_w  = (const float*)d_in[5];
    const float* dt_proj_w = (const float*)d_in[6];
    const float* dt_proj_b = (const float*)d_in[7];
    const float* A_log     = (const float*)d_in[8];
    const float* D_skip    = (const float*)d_in[9];
    const float* out_proj_w= (const float*)d_in[10];
    float* out = (float*)d_out;

    char* wsb = (char*)d_ws;
    size_t off = 0;
    auto alloc = [&](size_t bytes) -> void* {
        void* p = wsb + off;
        off += (bytes + 255) & ~(size_t)255;
        return p;
    };
    bf16_t* w_in  = (bf16_t*)alloc((size_t)2 * DI * DM * 2);
    bf16_t* w_dt  = (bf16_t*)alloc((size_t)DI * DI * 2);
    bf16_t* w_out = (bf16_t*)alloc((size_t)DM * DI * 2);
    bf16_t* w_xp  = (bf16_t*)alloc((size_t)32 * DI * 2);
    bf16_t* xn    = (bf16_t*)alloc((size_t)MTOT * DM * 2);
    float*  xz    = (float*) alloc((size_t)MTOT * 2 * DI * 4);
    float*  u     = (float*) alloc((size_t)MTOT * DI * 4);
    bf16_t* ubf   = (bf16_t*)alloc((size_t)MTOT * DI * 2);
    float*  dtraw = (float*) alloc((size_t)MTOT * DI * 4);
    float*  bc    = (float*) alloc((size_t)MTOT * 32 * 4);
    bf16_t* yg    = (bf16_t*)alloc((size_t)MTOT * DI * 2);
    if (off > ws_size) return;   // workspace too small: do nothing (deterministic)

    const int T = 256;
    auto cdiv = [](int a, int b) { return (a + b - 1) / b; };

    // 1) weights -> bf16
    mamba_cast_bf16<<<cdiv(2 * DI * DM, T), T, 0, stream>>>(in_proj_w,  w_in,  2 * DI * DM);
    mamba_cast_bf16<<<cdiv(DI * DI,     T), T, 0, stream>>>(dt_proj_w,  w_dt,  DI * DI);
    mamba_cast_bf16<<<cdiv(DM * DI,     T), T, 0, stream>>>(out_proj_w, w_out, DM * DI);
    mamba_cast_bf16<<<cdiv(32 * DI,     T), T, 0, stream>>>(x_proj_w,   w_xp,  32 * DI);

    // 2) RMSNorm -> bf16 activations
    mamba_rmsnorm<<<MTOT, T, 0, stream>>>(x, norm_w, xn);

    auto gemm = [&](const bf16_t* A, int lda, const bf16_t* Bw, int ldb,
                    float* C, int ldc, int M, int N, int K, const float* add) {
        int blocks = (M / BM) * cdiv(N, BN);
        mamba_gemm_wmma<<<blocks, 256, 0, stream>>>(A, lda, Bw, ldb, C, ldc, M, N, K, add);
    };

    // 3) xz = xn @ in_proj_w^T            (4096 x 3072 x 768)
    gemm(xn, DM, w_in, DM, xz, 2 * DI, MTOT, 2 * DI, DM, nullptr);

    // 4) depthwise causal conv + SiLU -> u (f32 + bf16)
    mamba_conv_silu<<<cdiv(MTOT * DI, T), T, 0, stream>>>(xz, conv_w, conv_b, u, ubf);

    // 5) bc = u @ x_proj_w^T               (4096 x 32 x 1536)
    gemm(ubf, DI, w_xp, DI, bc, 32, MTOT, 32, DI, nullptr);

    // 6) dtraw = u @ dt_proj_w^T           (4096 x 1536 x 1536)
    gemm(ubf, DI, w_dt, DI, dtraw, DI, MTOT, DI, DI, nullptr);

    // 7) selective scan + D-skip + silu(z) gating -> yg (bf16)
    mamba_scan<<<NB * (DI / 256), 256, 0, stream>>>(dtraw, dt_proj_b, u, xz, bc, A_log, D_skip, yg);

    // 8) out = yg @ out_proj_w^T + residual x   (4096 x 768 x 1536)
    gemm(yg, DI, w_out, DI, out, DM, MTOT, DM, DI, x);
}
